// CausalWanSelfAttention_50826642980887
// MI455X (gfx1250) — compile-verified, matrix operand order checked by reading, not run on hardware
//
#include <hip/hip_runtime.h>
#include <hip/hip_bf16.h>
#include <math.h>

#define NUM_HEADS 12
#define DIM 1536
#define HEAD_DIM 128
#define MAX_ATTN 5632
#define EPSV 1e-6f
// (1/sqrt(HEAD_DIM)) * log2(e): fold softmax scale + natural->base2 exp into Q.
#define QSCALE 0.1275174308f

typedef __attribute__((ext_vector_type(16))) __bf16 bf16x16;
typedef __attribute__((ext_vector_type(8)))  float  f32x8;

static __device__ inline f32x8 wmma_bf16(bf16x16 a, bf16x16 b, f32x8 c) {
  // D = A(16x32 bf16) * B(32x16 bf16) + C(16x16 f32)
  return __builtin_amdgcn_wmma_f32_16x16x32_bf16(false, a, false, b, (short)0, c,
                                                 false, false);
}

// ---------------------------------------------------------------- prep kernels

__global__ void cvt_bf16_kernel(const float* __restrict__ src,
                                __bf16* __restrict__ dst, int n) {
  int i = blockIdx.x * blockDim.x + threadIdx.x;
  if (i < n) dst[i] = (__bf16)src[i];
}

// cos/sin table per (seq position, rope pair j in [0,64)).
__global__ void rope_table_kernel(const float* __restrict__ theta,
                                  float* __restrict__ cosb, float* __restrict__ sinb,
                                  const int* p_f, const int* p_h, const int* p_w,
                                  const int* p_cs, int S) {
  int idx = blockIdx.x * blockDim.x + threadIdx.x;
  if (idx >= S * 64) return;
  const int s = idx >> 6;
  const int j = idx & 63;
  const int H  = p_h[0];
  const int Wd = p_w[0];
  const int fs = H * Wd;
  const int sf = p_cs[0] / fs;           // start_frame
  const int f   = s / fs;
  const int rem = s - f * fs;
  const int h   = rem / Wd;
  const int w   = rem - h * Wd;
  // c = 64 -> ct = 64 - 2*(64/3) = 22, ch = 21 (boundaries 22 and 43)
  float ang;
  if (j < 22)      ang = theta[(size_t)(sf + f) * 64 + j];
  else if (j < 43) ang = theta[(size_t)h * 64 + j];
  else             ang = theta[(size_t)w * 64 + j];
  cosb[idx] = cosf(ang);
  sinb[idx] = sinf(ang);
}

// Copy the cached prefix of the attention window into bf16 K (row-major) and
// V^T ([head][d][k]) buffers.
__global__ void cache_copy_kernel(const float* __restrict__ ck,
                                  const float* __restrict__ cv,
                                  __bf16* __restrict__ kwinb,
                                  __bf16* __restrict__ vwinT,
                                  const int* p_cs, const int* p_ge,
                                  const int* p_le, int S) {
  const int cs          = p_cs[0];
  const int local_end   = p_le[0] + (cs + S) - p_ge[0];
  const int local_start = local_end - S;
  const int win_start   = (local_end - MAX_ATTN) > 0 ? (local_end - MAX_ATTN) : 0;
  const int prefix      = local_start - win_start;   // rows sourced from cache
  const int stride = NUM_HEADS * HEAD_DIM;
  int idx = blockIdx.x * blockDim.x + threadIdx.x;
  if (idx >= prefix * stride) return;
  const int t = idx / stride;
  const int r = idx - t * stride;
  const int n = r / HEAD_DIM;
  const int d = r - n * HEAD_DIM;
  const size_t src = ((size_t)(win_start + t) * NUM_HEADS + n) * HEAD_DIM + d;
  kwinb[((size_t)n * MAX_ATTN + t) * HEAD_DIM + d] = (__bf16)ck[src];
  vwinT[((size_t)n * HEAD_DIM + d) * MAX_ATTN + t] = (__bf16)cv[src];
}

// ---------------------------------------------------------------- fused GEMM
// One workgroup = 32 rows x 1536 cols of one projection (16 waves, 512 thr).
// Each wave: 2 resident A tiles (32 rows) x 6 col-tiles (96 cols); every B
// operand load feeds two WMMAs.
// mode 0: Q (+bias, rmsnorm, rope, *QSCALE -> rqb bf16 [n][s][d])
// mode 1: K (+bias, rmsnorm, rope         -> kwinb bf16 [n][win][d])
// mode 2: V (+bias                         -> vwinT bf16 [n][d][win])
// mode 3: out-projection (+bias, f32      -> d_out [s][col])
__global__ __launch_bounds__(512) void gemm_kernel(
    const __bf16* __restrict__ A,
    const __bf16* __restrict__ Wq, const __bf16* __restrict__ Wk,
    const __bf16* __restrict__ Wv, const __bf16* __restrict__ Wo,
    const float* __restrict__ bq, const float* __restrict__ bk,
    const float* __restrict__ bv, const float* __restrict__ bo,
    const float* __restrict__ gq, const float* __restrict__ gk,
    const float* __restrict__ cosb, const float* __restrict__ sinb,
    __bf16* __restrict__ rqb, __bf16* __restrict__ kwinb,
    __bf16* __restrict__ vwinT, float* __restrict__ outf,
    const int* p_cs, const int* p_ge, const int* p_le,
    int S, int mode_base) {
  const int mode = mode_base + blockIdx.y;
  const int row0 = blockIdx.x * 32;
  const int wid  = threadIdx.x >> 5;     // 0..15
  const int lane = threadIdx.x & 31;
  const int lm   = lane & 15;
  const int lh   = lane >> 4;
  const int colbase = wid * 96;

  const __bf16* W   = (mode == 0) ? Wq : (mode == 1) ? Wk : (mode == 2) ? Wv : Wo;
  const float* bias = (mode == 0) ? bq : (mode == 1) ? bk : (mode == 2) ? bv : bo;

  f32x8 zero8 = {0.f, 0.f, 0.f, 0.f, 0.f, 0.f, 0.f, 0.f};
  f32x8 acc[2][6];
#pragma unroll
  for (int t = 0; t < 2; ++t)
#pragma unroll
    for (int ct = 0; ct < 6; ++ct) acc[t][ct] = zero8;

  // A operands: lane holds rows (row0 + lm) and (row0 + 16 + lm), clamped.
  int ar0 = row0 + lm;      if (ar0 >= S) ar0 = S - 1;
  int ar1 = row0 + 16 + lm; if (ar1 >= S) ar1 = S - 1;
  const __bf16* arow0 = A + (size_t)ar0 * DIM + lh * 16;
  const __bf16* arow1 = A + (size_t)ar1 * DIM + lh * 16;

  for (int kk = 0; kk < DIM; kk += 32) {
    bf16x16 a0 = *(const bf16x16*)(arow0 + kk);
    bf16x16 a1 = *(const bf16x16*)(arow1 + kk);
#pragma unroll
    for (int ct = 0; ct < 6; ++ct) {
      const __bf16* bp = W + (size_t)(colbase + ct * 16 + lm) * DIM + lh * 16 + kk;
      bf16x16 b = *(const bf16x16*)bp;
      acc[0][ct] = wmma_bf16(a0, b, acc[0][ct]);
      acc[1][ct] = wmma_bf16(a1, b, acc[1][ct]);
    }
  }

  // Bias (before rmsnorm, matching reference).
#pragma unroll
  for (int ct = 0; ct < 6; ++ct) {
    float b = bias[colbase + ct * 16 + lm];
#pragma unroll
    for (int t = 0; t < 2; ++t)
#pragma unroll
      for (int r = 0; r < 8; ++r) acc[t][ct][r] += b;
  }

  const int cs          = p_cs[0];
  const int local_end   = p_le[0] + (cs + S) - p_ge[0];
  const int local_start = local_end - S;
  const int win_start   = (local_end - MAX_ATTN) > 0 ? (local_end - MAX_ATTN) : 0;
  const int prefix      = local_start - win_start;

  __shared__ float ssq[32];

  if (mode <= 1) {
    if (threadIdx.x < 32) ssq[threadIdx.x] = 0.0f;
    __syncthreads();
    // Per-row partial sum of squares over this wave's 96 cols.
#pragma unroll
    for (int t = 0; t < 2; ++t) {
#pragma unroll
      for (int r = 0; r < 8; ++r) {
        float s = 0.f;
#pragma unroll
        for (int ct = 0; ct < 6; ++ct) s += acc[t][ct][r] * acc[t][ct][r];
#pragma unroll
        for (int m = 1; m < 16; m <<= 1) s += __shfl_xor(s, m, 32);
        if (lm == 0) atomicAdd(&ssq[t * 16 + r + 8 * lh], s);
      }
    }
    __syncthreads();
    const float* g = (mode == 0) ? gq : gk;
    float rms[2][8];
#pragma unroll
    for (int t = 0; t < 2; ++t)
#pragma unroll
      for (int r = 0; r < 8; ++r)
        rms[t][r] = rsqrtf(ssq[t * 16 + r + 8 * lh] * (1.0f / DIM) + EPSV);

#pragma unroll
    for (int ct = 0; ct < 6; ++ct) {
      const int col  = colbase + ct * 16 + lm;
      const float gg = g[col];
      const int hd   = col & (HEAD_DIM - 1);
      const int j    = hd >> 1;
      const int head = col >> 7;
#pragma unroll
      for (int t = 0; t < 2; ++t) {
#pragma unroll
        for (int r = 0; r < 8; ++r) {
          const int srow = row0 + t * 16 + r + 8 * lh;
          const bool ok  = srow < S;
          const int src  = ok ? srow : (S - 1);
          float y = acc[t][ct][r] * rms[t][r] * gg;
          // RoPE: real/imag pair sits in adjacent lanes (col parity == lane parity)
          float other = __shfl_xor(y, 1, 32);
          float cv = cosb[src * 64 + j];
          float sv = sinb[src * 64 + j];
          float out = (col & 1) ? (other * sv + y * cv) : (y * cv - other * sv);
          if (ok) {
            if (mode == 0) {
              rqb[((size_t)head * S + srow) * HEAD_DIM + hd] =
                  (__bf16)(out * QSCALE);
            } else {
              kwinb[((size_t)head * MAX_ATTN + prefix + srow) * HEAD_DIM + hd] =
                  (__bf16)out;
            }
          }
        }
      }
    }
  } else if (mode == 2) {
#pragma unroll
    for (int ct = 0; ct < 6; ++ct) {
      const int col  = colbase + ct * 16 + lm;
      const int hd   = col & (HEAD_DIM - 1);
      const int head = col >> 7;
#pragma unroll
      for (int t = 0; t < 2; ++t) {
#pragma unroll
        for (int r = 0; r < 8; ++r) {
          const int srow = row0 + t * 16 + r + 8 * lh;
          if (srow < S)
            vwinT[((size_t)head * HEAD_DIM + hd) * MAX_ATTN + prefix + srow] =
                (__bf16)acc[t][ct][r];
        }
      }
    }
  } else {
#pragma unroll
    for (int ct = 0; ct < 6; ++ct) {
      const int col = colbase + ct * 16 + lm;
#pragma unroll
      for (int t = 0; t < 2; ++t) {
#pragma unroll
        for (int r = 0; r < 8; ++r) {
          const int srow = row0 + t * 16 + r + 8 * lh;
          if (srow < S) outf[(size_t)srow * DIM + col] = acc[t][ct][r];
        }
      }
    }
  }
}

// ---------------------------------------------------------------- attention
// One wave per 32-query block per head (2 resident Q tiles); FA2 online
// softmax over the KV window. Each K/V B-operand load feeds two WMMAs.
__global__ __launch_bounds__(128) void attn_kernel(
    const __bf16* __restrict__ rqb, const __bf16* __restrict__ kwinb,
    const __bf16* __restrict__ vwinT, __bf16* __restrict__ ob,
    const int* p_cs, const int* p_ge, const int* p_le, int S) {
  __shared__ __attribute__((aligned(32))) __bf16 pbuf[4][2][16 * 32];
  const int head = blockIdx.y;
  const int wid  = threadIdx.x >> 5;   // 0..3
  const int lane = threadIdx.x & 31;
  const int lm   = lane & 15;
  const int lh   = lane >> 4;
  const int nqt  = S >> 4;
  const int qt0  = (blockIdx.x * 4 + wid) * 2;
  if (qt0 >= nqt) return;
  const bool q1ok = (qt0 + 1) < nqt;
  const int qtc[2] = {qt0, q1ok ? (qt0 + 1) : qt0};

  const int cs        = p_cs[0];
  const int local_end = p_le[0] + (cs + S) - p_ge[0];
  const int win_start = (local_end - MAX_ATTN) > 0 ? (local_end - MAX_ATTN) : 0;
  const int WLEN      = local_end - win_start;

  // Resident Q tiles: 2 x (16 x 128) bf16 as 8 A operands.
  bf16x16 aq[2][4];
#pragma unroll
  for (int qi = 0; qi < 2; ++qi) {
    const __bf16* qp =
        rqb + ((size_t)head * S + qtc[qi] * 16 + lm) * HEAD_DIM + lh * 16;
#pragma unroll
    for (int i = 0; i < 4; ++i) aq[qi][i] = *(const bf16x16*)(qp + i * 32);
  }

  f32x8 zero8 = {0.f, 0.f, 0.f, 0.f, 0.f, 0.f, 0.f, 0.f};
  f32x8 o[2][8];
  float rmax[2][8], rsum[2][8];
#pragma unroll
  for (int qi = 0; qi < 2; ++qi) {
#pragma unroll
    for (int i = 0; i < 8; ++i) o[qi][i] = zero8;
#pragma unroll
    for (int r = 0; r < 8; ++r) { rmax[qi][r] = -3.0e30f; rsum[qi][r] = 0.0f; }
  }

  const __bf16* kh = kwinb + (size_t)head * MAX_ATTN * HEAD_DIM;
  const __bf16* vh = vwinT + (size_t)head * HEAD_DIM * MAX_ATTN;

  const int nchunk = (WLEN + 31) >> 5;
  for (int c = 0; c < nchunk; ++c) {
    const int kb = c * 32;
    if (c + 1 < nchunk) {
      // Stage next chunk toward the caches (global_prefetch_b8).
      __builtin_prefetch(kh + (size_t)(kb + 32 + lane) * HEAD_DIM, 0, 1);
      __builtin_prefetch(vh + (size_t)(lane * 4) * MAX_ATTN + kb + 32, 0, 1);
    }
    // Scores: 2 q-tiles x 2 k-tiles; every K load feeds two WMMAs.
    f32x8 s[2][2];
#pragma unroll
    for (int qi = 0; qi < 2; ++qi) { s[qi][0] = zero8; s[qi][1] = zero8; }
#pragma unroll
    for (int ks = 0; ks < 4; ++ks) {
      bf16x16 b0 = *(const bf16x16*)(kh + (size_t)(kb + lm) * HEAD_DIM +
                                     ks * 32 + lh * 16);
      bf16x16 b1 = *(const bf16x16*)(kh + (size_t)(kb + 16 + lm) * HEAD_DIM +
                                     ks * 32 + lh * 16);
#pragma unroll
      for (int qi = 0; qi < 2; ++qi) {
        s[qi][0] = wmma_bf16(aq[qi][ks], b0, s[qi][0]);
        s[qi][1] = wmma_bf16(aq[qi][ks], b1, s[qi][1]);
      }
    }
    if (kb + 32 > WLEN) {  // tail mask
      const bool ok0 = (kb + lm) < WLEN;
      const bool ok1 = (kb + 16 + lm) < WLEN;
#pragma unroll
      for (int qi = 0; qi < 2; ++qi) {
#pragma unroll
        for (int r = 0; r < 8; ++r) {
          if (!ok0) s[qi][0][r] = -3.0e30f;
          if (!ok1) s[qi][1][r] = -3.0e30f;
        }
      }
    }
    // Online softmax (base-2). Row lives across 16 lanes of one half.
#pragma unroll
    for (int qi = 0; qi < 2; ++qi) {
      float alpha[8];
#pragma unroll
      for (int r = 0; r < 8; ++r) {
        float m = fmaxf(s[qi][0][r], s[qi][1][r]);
#pragma unroll
        for (int sh = 1; sh < 16; sh <<= 1) m = fmaxf(m, __shfl_xor(m, sh, 32));
        const float mn = fmaxf(rmax[qi][r], m);
        alpha[r] = exp2f(rmax[qi][r] - mn);
        rmax[qi][r] = mn;
        const float p0 = exp2f(s[qi][0][r] - mn);
        const float p1 = exp2f(s[qi][1][r] - mn);
        s[qi][0][r] = p0; s[qi][1][r] = p1;
        float ps = p0 + p1;
#pragma unroll
        for (int sh = 1; sh < 16; sh <<= 1) ps += __shfl_xor(ps, sh, 32);
        rsum[qi][r] = rsum[qi][r] * alpha[r] + ps;
      }
#pragma unroll
      for (int dt = 0; dt < 8; ++dt)
#pragma unroll
        for (int r = 0; r < 8; ++r) o[qi][dt][r] *= alpha[r];

      // P: C-layout -> A-layout via per-wave LDS staging.
      __bf16* pl = &pbuf[wid][qi][0];
#pragma unroll
      for (int r = 0; r < 8; ++r) {
        pl[(r + 8 * lh) * 32 + lm]      = (__bf16)s[qi][0][r];
        pl[(r + 8 * lh) * 32 + 16 + lm] = (__bf16)s[qi][1][r];
      }
    }
    asm volatile("s_wait_dscnt 0" ::: "memory");
    bf16x16 ap[2];
#pragma unroll
    for (int qi = 0; qi < 2; ++qi)
      ap[qi] = *(const bf16x16*)(&pbuf[wid][qi][0] + lm * 32 + lh * 16);
    asm volatile("" ::: "memory");

    // PV: B operand rows are head-dims, K-contiguous thanks to V^T layout;
    // every V load feeds two WMMAs.
#pragma unroll
    for (int dt = 0; dt < 8; ++dt) {
      bf16x16 bv = *(const bf16x16*)(vh + (size_t)(dt * 16 + lm) * MAX_ATTN +
                                     kb + lh * 16);
#pragma unroll
      for (int qi = 0; qi < 2; ++qi) o[qi][dt] = wmma_bf16(ap[qi], bv, o[qi][dt]);
    }
  }

#pragma unroll
  for (int qi = 0; qi < 2; ++qi) {
    if (qi == 1 && !q1ok) break;
    float inv[8];
#pragma unroll
    for (int r = 0; r < 8; ++r) inv[r] = 1.0f / rsum[qi][r];
#pragma unroll
    for (int dt = 0; dt < 8; ++dt) {
#pragma unroll
      for (int r = 0; r < 8; ++r) {
        const int srow = qtc[qi] * 16 + r + 8 * lh;
        ob[(size_t)srow * DIM + head * HEAD_DIM + dt * 16 + lm] =
            (__bf16)(o[qi][dt][r] * inv[r]);
      }
    }
  }
}

// ---------------------------------------------------------------- launch

extern "C" void kernel_launch(void* const* d_in, const int* in_sizes, int n_in,
                              void* d_out, int out_size, void* d_ws, size_t ws_size,
                              hipStream_t stream) {
  const float* x  = (const float*)d_in[0];
  const float* th = (const float*)d_in[1];
  const float* ck = (const float*)d_in[2];
  const float* cv = (const float*)d_in[3];
  const float* wq = (const float*)d_in[4];
  const float* bq = (const float*)d_in[5];
  const float* wk = (const float*)d_in[6];
  const float* bk = (const float*)d_in[7];
  const float* wv = (const float*)d_in[8];
  const float* bv = (const float*)d_in[9];
  const float* wo = (const float*)d_in[10];
  const float* bo = (const float*)d_in[11];
  const float* gq = (const float*)d_in[12];
  const float* gk = (const float*)d_in[13];
  const int* p_f  = (const int*)d_in[14];
  const int* p_h  = (const int*)d_in[15];
  const int* p_w  = (const int*)d_in[16];
  const int* p_cs = (const int*)d_in[17];
  const int* p_ge = (const int*)d_in[18];
  const int* p_le = (const int*)d_in[19];
  const int S = in_sizes[0] / DIM;

  char* wsp = (char*)d_ws;
  size_t off = 0;
  auto take = [&](size_t bytes) -> void* {
    void* p = wsp + off;
    off = (off + bytes + 255) & ~(size_t)255;
    return p;
  };
  __bf16* xb    = (__bf16*)take((size_t)S * DIM * 2);
  __bf16* wqb   = (__bf16*)take((size_t)DIM * DIM * 2);
  __bf16* wkb   = (__bf16*)take((size_t)DIM * DIM * 2);
  __bf16* wvb   = (__bf16*)take((size_t)DIM * DIM * 2);
  __bf16* wob   = (__bf16*)take((size_t)DIM * DIM * 2);
  float*  cosb  = (float*)take((size_t)S * 64 * 4);
  float*  sinb  = (float*)take((size_t)S * 64 * 4);
  __bf16* rqb   = (__bf16*)take((size_t)NUM_HEADS * S * HEAD_DIM * 2);
  __bf16* kwinb = (__bf16*)take((size_t)NUM_HEADS * MAX_ATTN * HEAD_DIM * 2);
  __bf16* vwinT = (__bf16*)take((size_t)NUM_HEADS * HEAD_DIM * MAX_ATTN * 2);
  __bf16* ob    = (__bf16*)take((size_t)S * DIM * 2);

  int n;
  n = S * DIM;
  cvt_bf16_kernel<<<(n + 255) / 256, 256, 0, stream>>>(x, xb, n);
  n = DIM * DIM;
  cvt_bf16_kernel<<<(n + 255) / 256, 256, 0, stream>>>(wq, wqb, n);
  cvt_bf16_kernel<<<(n + 255) / 256, 256, 0, stream>>>(wk, wkb, n);
  cvt_bf16_kernel<<<(n + 255) / 256, 256, 0, stream>>>(wv, wvb, n);
  cvt_bf16_kernel<<<(n + 255) / 256, 256, 0, stream>>>(wo, wob, n);
  n = S * 64;
  rope_table_kernel<<<(n + 255) / 256, 256, 0, stream>>>(th, cosb, sinb,
                                                         p_f, p_h, p_w, p_cs, S);
  n = MAX_ATTN * NUM_HEADS * HEAD_DIM;  // worst-case prefix; guarded in-kernel
  cache_copy_kernel<<<(n + 255) / 256, 256, 0, stream>>>(ck, cv, kwinb, vwinT,
                                                         p_cs, p_ge, p_le, S);

  const int nrow32 = (S + 31) / 32;
  dim3 gqkv(nrow32, 3);
  gemm_kernel<<<gqkv, 512, 0, stream>>>(xb, wqb, wkb, wvb, wob, bq, bk, bv, bo,
                                        gq, gk, cosb, sinb, rqb, kwinb, vwinT,
                                        (float*)d_out, p_cs, p_ge, p_le, S, 0);

  const int nqt  = S / 16;
  const int nq32 = (nqt + 1) / 2;
  dim3 gatt((nq32 + 3) / 4, NUM_HEADS);
  attn_kernel<<<gatt, 128, 0, stream>>>(rqb, kwinb, vwinT, ob,
                                        p_cs, p_ge, p_le, S);

  dim3 gout(nrow32, 1);
  gemm_kernel<<<gout, 512, 0, stream>>>(ob, wqb, wkb, wvb, wob, bq, bk, bv, bo,
                                        gq, gk, cosb, sinb, rqb, kwinb, vwinT,
                                        (float*)d_out, p_cs, p_ge, p_le, S, 3);
}
